// MultiHeadSelfAttention_41686952575158
// MI455X (gfx1250) — matrix operand order repaired and checked by reading, not compiled, not run on hardware
//
#include <hip/hip_runtime.h>
#include <hip/hip_bf16.h>

// MI455X / gfx1250 multi-head self-attention, bf16 WMMA pipeline + TDM staging.
// B=4, T=2048, C=1024, H=16, D=64.

#define BATCH 4
#define SEQ   2048
#define EMB   1024
#define HEADS 16
#define HDIM  64
#define MROWS (BATCH * SEQ)   // 8192

typedef __attribute__((ext_vector_type(16))) __bf16 v16bf;
typedef __attribute__((ext_vector_type(2)))  __bf16 v2bf;
typedef __attribute__((ext_vector_type(8)))  float  v8f;
typedef unsigned int uint32x4 __attribute__((ext_vector_type(4)));
typedef int          int32x8  __attribute__((ext_vector_type(8)));
typedef int          int32x4  __attribute__((ext_vector_type(4)));

#if defined(__gfx1250__) && __has_builtin(__builtin_amdgcn_tensor_load_to_lds)
#define USE_TDM 1
#else
#define USE_TDM 0
#endif

union Frag16 {
  v16bf v;
  unsigned int u32[8];
  unsigned short u16[16];
};

// A-matrix (16x32, 16-bit) per-lane K index for half h (0..15).
// ISA 7.12.2: lanes 0-15 row M=0..15; v0..3 hold K 0..7 (lanes<16) / 8..15
// (lanes>=16); v4..7 hold K 16..23 / 24..31.
__device__ __forceinline__ int ka_idx(int lane, int h) {
  int v = h >> 1;
  return ((v & 4) << 2) + (((lane >> 4) & 1) << 3) + ((v & 3) << 1) + (h & 1);
}
// B-matrix (32x16, 16-bit) per-lane K index: same K order as the A fragment
// (lanes<16 hold K 0..7,16..23; lanes>=16 hold K 8..15,24..31).
__device__ __forceinline__ int kb_idx(int lane, int h) {
  return ka_idx(lane, h);
}

__device__ __forceinline__ unsigned short f2bf(float f) {
  unsigned int u = __float_as_uint(f);
  u += 0x7FFFu + ((u >> 16) & 1u);   // round-to-nearest-even
  return (unsigned short)(u >> 16);
}

// Packed fp32x2 -> bf16x2 (single v_cvt_pk_bf16_f32 where available).
__device__ __forceinline__ unsigned int pk2bf(float lo, float hi) {
#if defined(__AMDGCN__) && __has_builtin(__builtin_amdgcn_cvt_pk_bf16_f32)
  union { v2bf v; unsigned int u; } cv;
  cv.v = __builtin_amdgcn_cvt_pk_bf16_f32(lo, hi);
  return cv.u;
#else
  return (unsigned int)f2bf(lo) | ((unsigned int)f2bf(hi) << 16);
#endif
}

__device__ __forceinline__ v8f wmma_bf16(const Frag16& a, const Frag16& b, v8f c) {
  // (neg_a, A, neg_b, B, c_mod, C, reuse_a, reuse_b)
  return __builtin_amdgcn_wmma_f32_16x16x32_bf16(false, a.v, false, b.v,
                                                 (short)0, c, false, false);
}

// ---------------------------------------------------------------------------
// GEMM: Out[M,N] = A[M,K] * W[K,N] + bias,  M=8192, N=K=1024.
// Block: 256 threads (8 waves). Block tile 128x64, wave tile 16x64, k-step 32.
// AMODE 0: A fp32 (x).           AMODE 1: A bf16 (attn out, row-major [M,K]).
// OMODE 0: bf16 out, [B,H,T,D].  OMODE 1: fp32 out, row-major [M,N].
// ---------------------------------------------------------------------------
template <int AMODE, int OMODE>
__global__ __launch_bounds__(256) void mha_gemm_kernel(
    const void* __restrict__ Ap, const float* __restrict__ W,
    const float* __restrict__ bias, void* __restrict__ Outp) {
  __shared__ __align__(16) unsigned short Alds[128 * 32];   // [row][k]
  __shared__ __align__(16) unsigned short Btlds[64 * 32];   // [col][k] (transposed)

  const int tid  = threadIdx.x;
  const int wave = tid >> 5;
  const int lane = tid & 31;
  const int rowbase = blockIdx.x * 128;
  const int colbase = blockIdx.y * 64;

  v8f acc[4];
#pragma unroll
  for (int j = 0; j < 4; ++j)
#pragma unroll
    for (int i = 0; i < 8; ++i) acc[j][i] = 0.0f;

  const int ar = tid >> 1;         // A-tile row 0..127
  const int ac = (tid & 1) << 4;   // A-tile col 0 or 16
  const int br = tid >> 3;         // W-tile k row 0..31
  const int bc = (tid & 7) << 3;   // W-tile col 0..56 step 8

  for (int k0 = 0; k0 < EMB; k0 += 32) {
    __syncthreads();
    // ---- stage A tile (128 x 32) as bf16 ----
    if (AMODE == 0) {
      const float* src = (const float*)Ap + (size_t)(rowbase + ar) * EMB + k0 + ac;
      __align__(16) unsigned int tmp[8];
#pragma unroll
      for (int i = 0; i < 4; ++i) {
        float4 f = ((const float4*)src)[i];
        tmp[2 * i + 0] = pk2bf(f.x, f.y);
        tmp[2 * i + 1] = pk2bf(f.z, f.w);
      }
      uint4* dst = (uint4*)&Alds[ar * 32 + ac];
      dst[0] = ((const uint4*)tmp)[0];
      dst[1] = ((const uint4*)tmp)[1];
    } else {
      const unsigned short* src =
          (const unsigned short*)Ap + (size_t)(rowbase + ar) * EMB + k0 + ac;
      uint4* dst = (uint4*)&Alds[ar * 32 + ac];
      dst[0] = ((const uint4*)src)[0];
      dst[1] = ((const uint4*)src)[1];
    }
    // ---- stage W tile (32 x 64) transposed into [col][k] bf16 ----
    {
      const float* src = W + (size_t)(k0 + br) * EMB + colbase + bc;
      float4 g0 = ((const float4*)src)[0];
      float4 g1 = ((const float4*)src)[1];
      Btlds[(bc + 0) * 32 + br] = f2bf(g0.x);
      Btlds[(bc + 1) * 32 + br] = f2bf(g0.y);
      Btlds[(bc + 2) * 32 + br] = f2bf(g0.z);
      Btlds[(bc + 3) * 32 + br] = f2bf(g0.w);
      Btlds[(bc + 4) * 32 + br] = f2bf(g1.x);
      Btlds[(bc + 5) * 32 + br] = f2bf(g1.y);
      Btlds[(bc + 6) * 32 + br] = f2bf(g1.z);
      Btlds[(bc + 7) * 32 + br] = f2bf(g1.w);
    }
    __syncthreads();

    // ---- compute: 1 A-frag x 4 B-frags -> 4 WMMAs ----
    const int m = lane & 15;
    Frag16 a;
#pragma unroll
    for (int t = 0; t < 8; ++t)
      a.u32[t] = *(const unsigned int*)&Alds[(wave * 16 + m) * 32 + ka_idx(lane, 2 * t)];
#pragma unroll
    for (int j = 0; j < 4; ++j) {
      Frag16 bfr;
#pragma unroll
      for (int t = 0; t < 8; ++t)
        bfr.u32[t] = *(const unsigned int*)&Btlds[(j * 16 + m) * 32 + kb_idx(lane, 2 * t)];
      acc[j] = wmma_bf16(a, bfr, acc[j]);
    }
  }

  // ---- epilogue: C-layout row = r + 8*(lane>=16), col = lane&15 ----
  const int hi = (lane >> 4) & 1;
  const int n0 = lane & 15;
#pragma unroll
  for (int r = 0; r < 8; ++r) {
    const int row = rowbase + wave * 16 + r + 8 * hi;
#pragma unroll
    for (int j = 0; j < 4; ++j) {
      const int col = colbase + j * 16 + n0;
      const float val = acc[j][r] + bias[col];
      if (OMODE == 0) {
        const int bb = row >> 11, tt = row & (SEQ - 1);
        const int hh = col >> 6,  dd = col & (HDIM - 1);
        ((unsigned short*)Outp)[(((size_t)(bb * HEADS + hh)) * SEQ + tt) * HDIM + dd] =
            f2bf(val);
      } else {
        ((float*)Outp)[(size_t)row * EMB + col] = val;
      }
    }
  }
}

// ---------------------------------------------------------------------------
// Flash attention: block = 8 waves, handles one (b,h) and 128 q rows.
// Each wave owns a 16-row q tile; KV processed 32 keys at a time via LDS.
// K tile staged by the Tensor Data Mover (1-D contiguous 4KB descriptor);
// V tile staged transposed through registers.
// Q/K/V bf16 in [B,H,T,D]; output bf16 in [B,T,H,D] (= [M,K] for out-proj).
// ---------------------------------------------------------------------------
__global__ __launch_bounds__(256) void mha_attn_kernel(
    const unsigned short* __restrict__ Q, const unsigned short* __restrict__ K,
    const unsigned short* __restrict__ V, unsigned short* __restrict__ O) {
  __shared__ __align__(16) unsigned short Klds[32 * 64];    // [key][d]
  __shared__ __align__(16) unsigned short Vtlds[64 * 32];   // [d][key]
  __shared__ __align__(16) unsigned short Plds[8 * 16 * 32];// per-wave [row][key]

  const int tid  = threadIdx.x;
  const int wave = tid >> 5;
  const int lane = tid & 31;
  const int bh   = blockIdx.x;          // b*H + h
  const int b    = bh >> 4;
  const int h    = bh & 15;
  const int qbase = blockIdx.y * 128 + wave * 16;

  const unsigned short* Qb = Q + (size_t)bh * SEQ * HDIM;
  const unsigned short* Kb = K + (size_t)bh * SEQ * HDIM;
  const unsigned short* Vb = V + (size_t)bh * SEQ * HDIM;

  const int m  = lane & 15;
  const int hi = (lane >> 4) & 1;

  // Q fragments (16 rows x 64 head-dim -> two 16x32 A fragments), held in regs.
  Frag16 qa0, qa1;
  {
    const unsigned int* qrow = (const unsigned int*)(Qb + (size_t)(qbase + m) * HDIM);
#pragma unroll
    for (int t = 0; t < 8; ++t) {
      const int c = ka_idx(lane, 2 * t);
      qa0.u32[t] = qrow[c >> 1];
      qa1.u32[t] = qrow[(c + 32) >> 1];
    }
  }

  v8f o[4];
#pragma unroll
  for (int j = 0; j < 4; ++j)
#pragma unroll
    for (int i = 0; i < 8; ++i) o[j][i] = 0.0f;
  float mrow[8], lrow[8];
#pragma unroll
  for (int r = 0; r < 8; ++r) { mrow[r] = -3.0e38f; lrow[r] = 0.0f; }

  unsigned short* pw = &Plds[wave * 16 * 32];
  const float scale = 0.125f;   // 1/sqrt(64)

  for (int kv = 0; kv < SEQ; kv += 32) {
    __syncthreads();

#if USE_TDM
    // ---- K tile via Tensor Data Mover: 1-D contiguous 4KB, 512 x 8B ----
    if (wave == 0) {
      const unsigned long long gaddr =
          (unsigned long long)(uintptr_t)(Kb + (size_t)kv * HDIM);
      const unsigned int laddr = (unsigned int)(uintptr_t)Klds;  // LDS offset
      uint32x4 g0;
      g0.x = 1u;                                   // count=1, user descriptor
      g0.y = laddr;                                // lds_addr
      g0.z = (unsigned int)gaddr;                  // global_addr[31:0]
      g0.w = (unsigned int)((gaddr >> 32) & 0x1FFFFFFu) | (2u << 30);  // type=2
      int32x8 g1;
      g1[0] = (int)(3u << 16);                     // data_size=8B
      g1[1] = (int)(512u << 16);                   // tensor_dim0 = 512
      g1[2] = (int)(1u << 16);                     // tensor_dim1 = 1
      g1[3] = (int)(512u << 16);                   // tile_dim0 = 512
      g1[4] = 1;                                   // tile_dim1 = 1, tile_dim2 = 0
      g1[5] = 512;                                 // tensor_dim0_stride = 512
      g1[6] = 0;
      g1[7] = 0;
      int32x4 z4 = {0, 0, 0, 0};
#if __clang_major__ >= 23
      int32x8 z8 = {0, 0, 0, 0, 0, 0, 0, 0};
      __builtin_amdgcn_tensor_load_to_lds(g0, g1, z4, z4, z8, 0);
#else
      __builtin_amdgcn_tensor_load_to_lds(g0, g1, z4, z4, 0);
#endif
#if __has_builtin(__builtin_amdgcn_s_wait_tensorcnt)
      __builtin_amdgcn_s_wait_tensorcnt(0);
#else
      asm volatile("s_wait_tensorcnt 0x0" ::: "memory");
#endif
    }
    asm volatile("" : : "r"(Klds) : "memory");  // TDM wrote Klds behind clang's back
#else
    // K tile: 32 keys x 64 d, contiguous 4KB -> straight b128 copy.
    ((uint4*)Klds)[tid] = ((const uint4*)(Kb + (size_t)kv * HDIM))[tid];
#endif

    // V tile: load 8 bf16 / thread, scatter transposed [d][key].
    {
      uint4 vv = ((const uint4*)(Vb + (size_t)kv * HDIM))[tid];
      const unsigned short* pv = (const unsigned short*)&vv;
      const int e0 = tid << 3, key = e0 >> 6, d0 = e0 & 63;
#pragma unroll
      for (int i = 0; i < 8; ++i) Vtlds[(d0 + i) * 32 + key] = pv[i];
    }
    if (kv + 32 < SEQ) {
      __builtin_prefetch(Kb + (size_t)(kv + 32) * HDIM, 0, 0);
      __builtin_prefetch(Vb + (size_t)(kv + 32) * HDIM, 0, 0);
    }
    __syncthreads();

    // ---- S = Q * K^T  (16 x 32 scores via two 16x16 accumulators) ----
    v8f S0, S1;
#pragma unroll
    for (int i = 0; i < 8; ++i) { S0[i] = 0.0f; S1[i] = 0.0f; }
    {
      Frag16 bk;
#pragma unroll
      for (int t = 0; t < 8; ++t)
        bk.u32[t] = *(const unsigned int*)&Klds[m * 64 + kb_idx(lane, 2 * t)];
      S0 = wmma_bf16(qa0, bk, S0);
#pragma unroll
      for (int t = 0; t < 8; ++t)
        bk.u32[t] = *(const unsigned int*)&Klds[m * 64 + 32 + kb_idx(lane, 2 * t)];
      S0 = wmma_bf16(qa1, bk, S0);
#pragma unroll
      for (int t = 0; t < 8; ++t)
        bk.u32[t] = *(const unsigned int*)&Klds[(m + 16) * 64 + kb_idx(lane, 2 * t)];
      S1 = wmma_bf16(qa0, bk, S1);
#pragma unroll
      for (int t = 0; t < 8; ++t)
        bk.u32[t] = *(const unsigned int*)&Klds[(m + 16) * 64 + 32 + kb_idx(lane, 2 * t)];
      S1 = wmma_bf16(qa1, bk, S1);
    }

    // ---- online softmax (f32); rows live on 16-lane groups ----
#pragma unroll
    for (int r = 0; r < 8; ++r) {
      float s0 = S0[r] * scale;
      float s1 = S1[r] * scale;
      float mx = fmaxf(s0, s1);
      mx = fmaxf(mx, __shfl_xor(mx, 1, 32));
      mx = fmaxf(mx, __shfl_xor(mx, 2, 32));
      mx = fmaxf(mx, __shfl_xor(mx, 4, 32));
      mx = fmaxf(mx, __shfl_xor(mx, 8, 32));
      const float mnew = fmaxf(mrow[r], mx);
      const float corr = __expf(mrow[r] - mnew);
      mrow[r] = mnew;
      const float p0 = __expf(s0 - mnew);
      const float p1 = __expf(s1 - mnew);
      float rs = p0 + p1;
      rs += __shfl_xor(rs, 1, 32);
      rs += __shfl_xor(rs, 2, 32);
      rs += __shfl_xor(rs, 4, 32);
      rs += __shfl_xor(rs, 8, 32);
      lrow[r] = lrow[r] * corr + rs;
#pragma unroll
      for (int j = 0; j < 4; ++j) o[j][r] *= corr;
      // stage P (bf16) for re-swizzle C-layout -> A-layout (same wave only)
      pw[(r + 8 * hi) * 32 + m]      = f2bf(p0);
      pw[(r + 8 * hi) * 32 + 16 + m] = f2bf(p1);
    }

    // ---- O += P * V  (P: 16x32 A-frag from LDS; V: 4 B-frags 32x16) ----
    Frag16 pa;
#pragma unroll
    for (int t = 0; t < 8; ++t)
      pa.u32[t] = *(const unsigned int*)&pw[m * 32 + ka_idx(lane, 2 * t)];
#pragma unroll
    for (int j = 0; j < 4; ++j) {
      Frag16 bv;
#pragma unroll
      for (int t = 0; t < 8; ++t)
        bv.u32[t] = *(const unsigned int*)&Vtlds[(j * 16 + m) * 32 + kb_idx(lane, 2 * t)];
      o[j] = wmma_bf16(pa, bv, o[j]);
    }
  }

  // ---- normalize and store bf16 [B,T,H,D] ----
#pragma unroll
  for (int r = 0; r < 8; ++r) {
    const float inv = 1.0f / lrow[r];
    const int row = qbase + r + 8 * hi;
#pragma unroll
    for (int j = 0; j < 4; ++j) {
      const int d = j * 16 + m;
      O[(((size_t)b * SEQ + row) * HEADS + h) * HDIM + d] = f2bf(o[j][r] * inv);
    }
  }
}

// ---------------------------------------------------------------------------
extern "C" void kernel_launch(void* const* d_in, const int* in_sizes, int n_in,
                              void* d_out, int out_size, void* d_ws, size_t ws_size,
                              hipStream_t stream) {
  (void)in_sizes; (void)n_in; (void)out_size; (void)ws_size;
  const float* x  = (const float*)d_in[0];
  const float* Wq = (const float*)d_in[1];
  const float* bq = (const float*)d_in[2];
  const float* Wk = (const float*)d_in[3];
  const float* bk = (const float*)d_in[4];
  const float* Wv = (const float*)d_in[5];
  const float* bv = (const float*)d_in[6];
  const float* Wo = (const float*)d_in[7];
  const float* bo = (const float*)d_in[8];
  float* out = (float*)d_out;

  // Workspace: Q,K,V bf16 [B,H,T,D] + attn-out bf16 [B,T,H,D]  (4 x 16MB)
  const size_t NE = (size_t)MROWS * EMB;   // 8,388,608 elements
  unsigned short* Qws  = (unsigned short*)d_ws;
  unsigned short* Kws  = Qws + NE;
  unsigned short* Vws  = Kws + NE;
  unsigned short* AOws = Vws + NE;

  dim3 block(256);
  dim3 ggrid(MROWS / 128, EMB / 64);        // 64 x 16
  dim3 agrid(BATCH * HEADS, SEQ / 128);     // 64 x 16

  mha_gemm_kernel<0, 0><<<ggrid, block, 0, stream>>>(x, Wq, bq, Qws);
  mha_gemm_kernel<0, 0><<<ggrid, block, 0, stream>>>(x, Wk, bk, Kws);
  mha_gemm_kernel<0, 0><<<ggrid, block, 0, stream>>>(x, Wv, bv, Vws);
  mha_attn_kernel<<<agrid, block, 0, stream>>>(Qws, Kws, Vws, AOws);
  mha_gemm_kernel<1, 1><<<ggrid, block, 0, stream>>>(AOws, Wo, bo, out);
}
